// DualCNNLSTM_86543591014600
// MI455X (gfx1250) — matrix operand order repaired and checked by reading, not compiled, not run on hardware
//
#include <hip/hip_runtime.h>
#include <hip/hip_bf16.h>

// ---------------------------------------------------------------------------
// CDNA5 (gfx1250, wave32) implementation.
// Matmuls use v_wmma_f32_16x16x32_f16 (f16 A/B, f32 accum, K padded to 32).
// ---------------------------------------------------------------------------

typedef __attribute__((ext_vector_type(16))) _Float16 v16h;
typedef __attribute__((ext_vector_type(8)))  float    v8f;

#define Bn   128
#define Tn   224
#define Ln   224
#define Hn   12
#define ROWS (Bn * Tn)        // 28672

// ---- WMMA fragment loaders -------------------------------------------------
// A: row-major [M][Kpad] f16, 16x32 tile. Lane = M%16; the 16 halves are
// K = koff..koff+15 with koff = 16*(lane>=16) (matches load_frag_b).
__device__ __forceinline__ v16h load_frag_a(const _Float16* base, int ld, int lane) {
    int m    = lane & 15;
    int koff = (lane >> 4) << 4;          // 0 or 16
    const _Float16* p = base + m * ld;
    v16h r;
    ((float4*)&r)[0] = *(const float4*)(p + koff);
    ((float4*)&r)[1] = *(const float4*)(p + 8 + koff);
    return r;
}
// B: weights stored [N][Kpad] f16 (i.e. B^T), 32x16 tile. Lane = N%16; the 16
// halves are K = kb..kb+15 with kb = 16*(lane>=16) -> one contiguous 32B load.
__device__ __forceinline__ v16h load_frag_b(const _Float16* base, int ld, int lane) {
    int n  = lane & 15;
    int kb = lane & 16;                   // 0 or 16
    const _Float16* p = base + n * ld + kb;
    v16h r;
    ((float4*)&r)[0] = ((const float4*)p)[0];
    ((float4*)&r)[1] = ((const float4*)p)[1];
    return r;
}

__device__ __forceinline__ float sigf(float x) { return 1.f / (1.f + __expf(-x)); }

// ---------------------------------------------------------------------------
// 1) Conv chain: conv1d(k=8,s=4,p=2)+relu -> maxpool(4,s=2,p=1)+relu ->
//    conv1d(k=3,s=1,p=1)+relu.  Output padded to K=32 f16 rows (GEMM A).
// ---------------------------------------------------------------------------
__global__ void conv_chain_kernel(const float* __restrict__ x,
                                  const float* c1w1, const float* c1b1,
                                  const float* c1w2, const float* c1b2,
                                  const float* c2w1, const float* c2b1,
                                  const float* c2w2, const float* c2b2,
                                  _Float16* __restrict__ Yp)   // [2][ROWS][32]
{
    long id = (long)blockIdx.x * blockDim.x + threadIdx.x;
    if (id >= 2L * ROWS * 32) return;
    int  j   = (int)(id & 31);
    long row = id >> 5;                         // 0 .. 2*ROWS-1
    _Float16* out = Yp + row * 32;
    if (j >= 28) { out[j] = (_Float16)0.f; return; }   // K padding
    int s = row >= ROWS;
    int r = (int)(row - (long)s * ROWS);        // r = b*T + t
    const float* w1 = s ? c2w1 : c1w1;  const float b1 = s ? c2b1[0] : c1b1[0];
    const float* w2 = s ? c2w2 : c1w2;  const float b2 = s ? c2b2[0] : c1b2[0];
    const float* xr = x + ((long)(s * Bn + r / Tn)) * (Tn * Ln) + (long)(r % Tn) * Ln;

    float acc = b2;
    for (int k2 = 0; k2 < 3; ++k2) {
        int q = j + k2 - 1;                     // pooled index (conv2 zero-pad)
        float pooled = 0.f;
        if (q >= 0 && q < 28) {
            float mx = -1e30f;
            for (int k1 = 0; k1 < 4; ++k1) {
                int i = 2 * q - 1 + k1;         // conv1 output index (-inf pad)
                if (i < 0 || i >= 56) continue;
                float a = b1;
                for (int k0 = 0; k0 < 8; ++k0) {
                    int xi = 4 * i + k0 - 2;    // conv1 zero-pad
                    if (xi >= 0 && xi < Ln) a += xr[xi] * w1[k0];
                }
                a = fmaxf(a, 0.f);
                mx = fmaxf(mx, a);
            }
            pooled = fmaxf(mx, 0.f);
        }
        acc += pooled * w2[k2];
    }
    out[j] = (_Float16)fmaxf(acc, 0.f);
}

// ---------------------------------------------------------------------------
// 2) Weight packing: fp32 -> f16, LSTM weights into [N][Kpad=32] (B^T) layout.
// ---------------------------------------------------------------------------
__global__ void cvt_f16_kernel(const float* __restrict__ in, _Float16* __restrict__ out, int n) {
    int id = blockIdx.x * blockDim.x + threadIdx.x;
    if (id < n) out[id] = (_Float16)in[id];
}

__global__ void pack_lstm_kernel(const float* w10i, const float* w10h,
                                 const float* w11i, const float* w11h,
                                 const float* w20i, const float* w20h,
                                 const float* w21i, const float* w21h,
                                 _Float16* __restrict__ WihP,   // [4][96][32]
                                 _Float16* __restrict__ WhhP)   // [4][96][32] (=[4][2][48][32])
{
    int id = blockIdx.x * blockDim.x + threadIdx.x;
    const int NIH = 4 * 96 * 32;
    if (id < NIH) {
        int k = id & 31, n96 = (id >> 5) % 96, sl = id / (96 * 32);
        const float* src = (sl == 0) ? w10i : (sl == 1) ? w11i : (sl == 2) ? w20i : w21i;
        int din = (sl & 1) ? 24 : 28;
        float v = (k < din) ? src[n96 * din + k] : 0.f;   // n96 = d*48+n, Wih flat (2,48,din)
        WihP[id] = (_Float16)v;
    } else if (id < 2 * NIH) {
        int id2 = id - NIH;
        int k = id2 & 31, n96 = (id2 >> 5) % 96, sl = id2 / (96 * 32);
        const float* src = (sl == 0) ? w10h : (sl == 1) ? w11h : (sl == 2) ? w20h : w21h;
        float v = (k < Hn) ? src[n96 * Hn + k] : 0.f;
        WhhP[id2] = (_Float16)v;
    }
}

// ---------------------------------------------------------------------------
// 3) Generic WMMA GEMM: C[M][N] = A[M][K] * Bw[N][K]^T (+bias).  One wave per
//    16x16 output tile, K-loop in steps of 32.  f32 and/or f16 outputs.
// ---------------------------------------------------------------------------
__global__ void gemm_wmma_f16(const _Float16* __restrict__ A, int lda,
                              const _Float16* __restrict__ Bw, int ldb,
                              const float* __restrict__ bias,
                              float* __restrict__ Cf, _Float16* __restrict__ Ch, int ldc,
                              int M, int N, int K)
{
    int lane = threadIdx.x & 31;
    int wid  = blockIdx.x * (blockDim.x >> 5) + (threadIdx.x >> 5);
    int Nt   = N >> 4;
    int total = (M >> 4) * Nt;
    if (wid >= total) return;                       // wave-uniform: EXEC stays all-1
    int mt = wid / Nt, nt = wid % Nt;
    const _Float16* Ab = A + (long)(mt * 16) * lda;
    const _Float16* Bb = Bw + (long)(nt * 16) * ldb;

    v8f acc = {};
    for (int k0 = 0; k0 < K; k0 += 32) {
        if (k0 + 32 < K) {
            __builtin_prefetch(Ab + k0 + 32, 0, 1);  // global_prefetch_b8
            __builtin_prefetch(Bb + k0 + 32, 0, 1);
        }
        v16h a = load_frag_a(Ab + k0, lda, lane);
        v16h b = load_frag_b(Bb + k0, ldb, lane);
        acc = __builtin_amdgcn_wmma_f32_16x16x32_f16(false, a, false, b,
                                                     (short)0, acc, false, false);
    }
    int n     = nt * 16 + (lane & 15);
    int mbase = mt * 16 + ((lane >> 4) << 3);       // D layout: M = r + 8*(lane>=16)
    float bv = bias ? bias[n] : 0.f;
    for (int r = 0; r < 8; ++r) {
        float v = acc[r] + bv;
        long  o = (long)(mbase + r) * ldc + n;
        if (Cf) Cf[o] = v;
        if (Ch) Ch[o] = (_Float16)v;
    }
}

// ---------------------------------------------------------------------------
// 4) Persistent bidirectional LSTM scan.  grid = 2 blocks (dir), 8 waves/block.
//    Wave w owns batch rows 16w..16w+15 for the whole sequence: per step it does
//    3 WMMAs (h 16x32 * Whh^T 32x48), dumps gates to its private LDS tile,
//    pointwise-updates c/h, re-stages h as next step's A fragment. No barriers.
//    mode 0: write h concat as f16 [ROWS][32] (layer-0 out); mode 1: f32 [ROWS][24].
// ---------------------------------------------------------------------------
__global__ void lstm_scan_kernel(const float* __restrict__ GX,     // [ROWS][96]
                                 const float* __restrict__ bias,   // [2][48]
                                 const _Float16* __restrict__ Whh, // [2][48][32]
                                 _Float16* __restrict__ outH,      // mode 0
                                 float* __restrict__ outF,         // mode 1
                                 int mode)
{
    int d    = blockIdx.x;            // 0 fwd, 1 bwd
    int wv   = threadIdx.x >> 5;
    int lane = threadIdx.x & 31;

    __shared__ _Float16 hbuf[8][16 * 32];   // h staging, K-padded (cols 12..31 == 0)
    __shared__ float    gbuf[8][16 * 48];   // gate pre-activations
    __shared__ float    cbuf[8][16 * 12];   // cell state
    _Float16* hb = hbuf[wv];
    float*    gb = gbuf[wv];
    float*    cb = cbuf[wv];
    for (int i = lane; i < 16 * 32; i += 32) hb[i] = (_Float16)0.f;
    for (int i = lane; i < 16 * 12; i += 32) cb[i] = 0.f;

    const _Float16* Wd = Whh + d * 48 * 32;
    v16h b0 = load_frag_b(Wd,            32, lane);   // gates n 0..15
    v16h b1 = load_frag_b(Wd + 16 * 32,  32, lane);   // gates n 16..31
    v16h b2 = load_frag_b(Wd + 32 * 32,  32, lane);   // gates n 32..47
    const float* bvec = bias + d * 48;
    int mbase = wv * 16;                               // batch rows of this wave
    int r16 = (lane >> 4) << 3, nn = lane & 15;

    for (int step = 0; step < Tn; ++step) {
        int t = d ? (Tn - 1 - step) : step;
        v16h a = load_frag_a(hb, 32, lane);
        v8f g0 = {}, g1 = {}, g2 = {};
        g0 = __builtin_amdgcn_wmma_f32_16x16x32_f16(false, a, false, b0, (short)0, g0, false, false);
        g1 = __builtin_amdgcn_wmma_f32_16x16x32_f16(false, a, false, b1, (short)0, g1, false, false);
        g2 = __builtin_amdgcn_wmma_f32_16x16x32_f16(false, a, false, b2, (short)0, g2, false, false);
        for (int r = 0; r < 8; ++r) {
            int m = r + r16;
            gb[m * 48 +      nn] = g0[r];
            gb[m * 48 + 16 + nn] = g1[r];
            gb[m * 48 + 32 + nn] = g2[r];
        }
        // pointwise: 16 rows x 12 hidden = 192 elems, 6 per lane (same wave: LDS in-order)
        for (int e = lane; e < 16 * Hn; e += 32) {
            int m = e / Hn, j = e % Hn;
            long row = (long)(mbase + m) * Tn + t;
            const float* gx = GX + row * 96 + d * 48;
            float iv = gb[m * 48 +      j] + gx[     j] + bvec[     j];
            float fv = gb[m * 48 + 12 + j] + gx[12 + j] + bvec[12 + j];
            float gv = gb[m * 48 + 24 + j] + gx[24 + j] + bvec[24 + j];
            float ov = gb[m * 48 + 36 + j] + gx[36 + j] + bvec[36 + j];
            float c = sigf(fv) * cb[e] + sigf(iv) * tanhf(gv);
            cb[e] = c;
            float h = sigf(ov) * tanhf(c);
            hb[m * 32 + j] = (_Float16)h;
            if (mode == 0) outH[row * 32 + d * Hn + j] = (_Float16)h;
            else           outF[row * 24 + d * Hn + j] = h;
        }
        if (mode == 0 && d == 0) {                    // zero K-pad cols 24..31
            for (int e = lane; e < 16 * 8; e += 32) {
                int m = e / 8, j = 24 + (e % 8);
                long row = (long)(mbase + m) * Tn + t;
                outH[row * 32 + j] = (_Float16)0.f;
            }
        }
    }
}

// ---------------------------------------------------------------------------
// 5) Combine streams (contiguous: [b][t*24+j] == [b*T+t][j]) -> f16 FC input.
// ---------------------------------------------------------------------------
__global__ void combine_kernel(const float* __restrict__ O1, const float* __restrict__ O2,
                               const float* w1, const float* w2,
                               _Float16* __restrict__ OC)     // [128][5376]
{
    int id = blockIdx.x * blockDim.x + threadIdx.x;
    if (id >= Bn * Tn * 24) return;
    OC[id] = (_Float16)(w1[0] * O1[id] + w2[0] * O2[id]);
}

// ---------------------------------------------------------------------------
// 6) FC3 (N=9: too small for WMMA) -> d_out fp32 [128][9].
// ---------------------------------------------------------------------------
__global__ void fc3_kernel(const float* __restrict__ F2, const float* __restrict__ W,
                           const float* __restrict__ bias, float* __restrict__ out)
{
    int id = blockIdx.x * blockDim.x + threadIdx.x;
    if (id >= Bn * 9) return;
    int c = id % 9, b = id / 9;
    float a = bias[c];
    for (int k = 0; k < 192; ++k) a += F2[b * 192 + k] * W[c * 192 + k];
    out[id] = a;
}

// ---------------------------------------------------------------------------
extern "C" void kernel_launch(void* const* d_in, const int* in_sizes, int n_in,
                              void* d_out, int out_size, void* d_ws, size_t ws_size,
                              hipStream_t stream)
{
    const float* x     = (const float*)d_in[0];
    const float* c1w1  = (const float*)d_in[1];
    const float* c1b1  = (const float*)d_in[2];
    const float* c1w2  = (const float*)d_in[3];
    const float* c1b2  = (const float*)d_in[4];
    const float* c2w1  = (const float*)d_in[5];
    const float* c2b1  = (const float*)d_in[6];
    const float* c2w2  = (const float*)d_in[7];
    const float* c2b2  = (const float*)d_in[8];
    const float* l10Wi = (const float*)d_in[9];
    const float* l10Wh = (const float*)d_in[10];
    const float* l10b  = (const float*)d_in[11];
    const float* l11Wi = (const float*)d_in[12];
    const float* l11Wh = (const float*)d_in[13];
    const float* l11b  = (const float*)d_in[14];
    const float* l20Wi = (const float*)d_in[15];
    const float* l20Wh = (const float*)d_in[16];
    const float* l20b  = (const float*)d_in[17];
    const float* l21Wi = (const float*)d_in[18];
    const float* l21Wh = (const float*)d_in[19];
    const float* l21b  = (const float*)d_in[20];
    const float* fcw1  = (const float*)d_in[21];
    const float* fcb1  = (const float*)d_in[22];
    const float* fcw2  = (const float*)d_in[23];
    const float* fcb2  = (const float*)d_in[24];
    const float* fcw3  = (const float*)d_in[25];
    const float* fcb3  = (const float*)d_in[26];
    const float* w1s   = (const float*)d_in[27];
    const float* w2s   = (const float*)d_in[28];
    float* out = (float*)d_out;

    // ---- workspace layout ----
    char*  ws  = (char*)d_ws;
    size_t off = 0;
    auto alloc = [&](size_t bytes) -> void* {
        void* p = ws + off;
        off = (off + bytes + 255) & ~(size_t)255;
        return p;
    };
    _Float16* Yp    = (_Float16*)alloc(2L * ROWS * 32 * 2);    // conv out, f16, K-pad
    float*    GX0   = (float*)   alloc(2L * ROWS * 96 * 4);    // layer-0 gate precompute
    _Float16* H1    = (_Float16*)alloc(2L * ROWS * 32 * 2);    // layer-0 h concat, K-pad
    float*    GX1   = (float*)   alloc(2L * ROWS * 96 * 4);    // layer-1 gate precompute
    float*    Ow    = (float*)   alloc(2L * ROWS * 24 * 4);    // layer-1 h concat, f32
    _Float16* OC    = (_Float16*)alloc((size_t)Bn * 5376 * 2); // combined FC input
    _Float16* fcw1h = (_Float16*)alloc(1024L * 5376 * 2);
    _Float16* fcw2h = (_Float16*)alloc(192L * 1024 * 2);
    float*    F1f   = (float*)   alloc((size_t)Bn * 1024 * 4);
    _Float16* F1h   = (_Float16*)alloc((size_t)Bn * 1024 * 2);
    float*    F2f   = (float*)   alloc((size_t)Bn * 192 * 4);
    _Float16* WihP  = (_Float16*)alloc(4L * 96 * 32 * 2);      // [s*2+layer][96][32]
    _Float16* WhhP  = (_Float16*)alloc(4L * 96 * 32 * 2);      // [s*2+layer][2][48][32]
    (void)ws_size; (void)in_sizes; (void)n_in; (void)out_size;

    const int BLK = 256;

    // 1) conv chains (both streams)
    {
        long n = 2L * ROWS * 32;
        conv_chain_kernel<<<(int)((n + BLK - 1) / BLK), BLK, 0, stream>>>(
            x, c1w1, c1b1, c1w2, c1b2, c2w1, c2b1, c2w2, c2b2, Yp);
    }
    // 2) weight packing
    pack_lstm_kernel<<<(2 * 4 * 96 * 32 + BLK - 1) / BLK, BLK, 0, stream>>>(
        l10Wi, l10Wh, l11Wi, l11Wh, l20Wi, l20Wh, l21Wi, l21Wh, WihP, WhhP);
    cvt_f16_kernel<<<(1024 * 5376 + BLK - 1) / BLK, BLK, 0, stream>>>(fcw1, fcw1h, 1024 * 5376);
    cvt_f16_kernel<<<(192 * 1024 + BLK - 1) / BLK, BLK, 0, stream>>>(fcw2, fcw2h, 192 * 1024);

    // 3) layer-0 gate precompute GEMMs: [ROWS,32] x [96,32]^T -> [ROWS,96]
    {
        int waves = (ROWS / 16) * (96 / 16);
        int grid  = (waves + 7) / 8;
        gemm_wmma_f16<<<grid, BLK, 0, stream>>>(Yp,            32, WihP,              32,
                                                nullptr, GX0,              nullptr, 96,
                                                ROWS, 96, 32);
        gemm_wmma_f16<<<grid, BLK, 0, stream>>>(Yp + (long)ROWS * 32, 32, WihP + 2 * 96 * 32, 32,
                                                nullptr, GX0 + (long)ROWS * 96, nullptr, 96,
                                                ROWS, 96, 32);
    }
    // 4) layer-0 scans (per stream; grid.x = direction)
    lstm_scan_kernel<<<2, BLK, 0, stream>>>(GX0,                    l10b, WhhP,
                                            H1,                     nullptr, 0);
    lstm_scan_kernel<<<2, BLK, 0, stream>>>(GX0 + (long)ROWS * 96,  l20b, WhhP + 2 * 96 * 32,
                                            H1 + (long)ROWS * 32,   nullptr, 0);
    // 5) layer-1 gate precompute GEMMs
    {
        int waves = (ROWS / 16) * (96 / 16);
        int grid  = (waves + 7) / 8;
        gemm_wmma_f16<<<grid, BLK, 0, stream>>>(H1,                    32, WihP + 1 * 96 * 32, 32,
                                                nullptr, GX1,              nullptr, 96,
                                                ROWS, 96, 32);
        gemm_wmma_f16<<<grid, BLK, 0, stream>>>(H1 + (long)ROWS * 32,  32, WihP + 3 * 96 * 32, 32,
                                                nullptr, GX1 + (long)ROWS * 96, nullptr, 96,
                                                ROWS, 96, 32);
    }
    // 6) layer-1 scans
    lstm_scan_kernel<<<2, BLK, 0, stream>>>(GX1,                   l11b, WhhP + 1 * 96 * 32,
                                            nullptr, Ow,                    1);
    lstm_scan_kernel<<<2, BLK, 0, stream>>>(GX1 + (long)ROWS * 96, l21b, WhhP + 3 * 96 * 32,
                                            nullptr, Ow + (long)ROWS * 24,  1);
    // 7) combine streams -> f16 FC input [128][5376]
    combine_kernel<<<(Bn * Tn * 24 + BLK - 1) / BLK, BLK, 0, stream>>>(
        Ow, Ow + (long)ROWS * 24, w1s, w2s, OC);
    // 8) FC1: [128,5376] x [1024,5376]^T + b
    {
        int waves = (Bn / 16) * (1024 / 16);
        gemm_wmma_f16<<<(waves + 7) / 8, BLK, 0, stream>>>(OC, 5376, fcw1h, 5376,
                                                           fcb1, F1f, F1h, 1024,
                                                           Bn, 1024, 5376);
    }
    // 9) FC2: [128,1024] x [192,1024]^T + b
    {
        int waves = (Bn / 16) * (192 / 16);
        gemm_wmma_f16<<<(waves + 7) / 8, BLK, 0, stream>>>(F1h, 1024, fcw2h, 1024,
                                                           fcb2, F2f, nullptr, 192,
                                                           Bn, 192, 1024);
    }
    // 10) FC3 (tiny) -> d_out
    fc3_kernel<<<(Bn * 9 + BLK - 1) / BLK, BLK, 0, stream>>>(F2f, fcw3, fcb3, out);
}
